// GatedCBEASPP_23699629540093
// MI455X (gfx1250) — matrix operand order repaired and checked by reading, not compiled, not run on hardware
//
#include <hip/hip_runtime.h>
#include <math.h>

typedef __attribute__((ext_vector_type(16))) _Float16 v16h;
typedef __attribute__((ext_vector_type(8)))  float    v8f;
typedef __attribute__((ext_vector_type(4)))  unsigned int v4u;
typedef __attribute__((ext_vector_type(8)))  int      v8i;
typedef __attribute__((ext_vector_type(4)))  int      v4i;

#define WMMA_F16(A,B,C) __builtin_amdgcn_wmma_f32_16x16x32_f16(false,(A),false,(B),(short)0,(C),false,false)

#if defined(__AMDGCN__) && __has_builtin(__builtin_amdgcn_tensor_load_to_lds) && __has_builtin(__builtin_amdgcn_s_wait_tensorcnt)
#define HAVE_TDM 1
#else
#define HAVE_TDM 0
#endif

__device__ __forceinline__ float sigmf(float x){
#if defined(__AMDGCN__)
  return __builtin_amdgcn_rcpf(1.f + __expf(-x));   // v_rcp_f32, no div chain
#else
  return 1.f/(1.f + __expf(-x));
#endif
}

#if HAVE_TDM
// Issue one TDM 3D-tile load: X contiguous (tile_x elems, 2B each), Y stride sy elems,
// Z stride sz elems. LDS receives tile fully packed: [Z][Y][X].
__device__ __forceinline__ void tdm_load_tile(
    unsigned lds_byte_addr, const void* gsrc,
    unsigned tile_x, unsigned tile_y, unsigned tile_z,
    unsigned long long sy, unsigned long long sz,
    unsigned tdim0, unsigned tdim1, unsigned tdim2)
{
  unsigned long long ga = (unsigned long long)(size_t)gsrc;
  v4u g0;
  g0[0] = 1u;                                              // count=1, user D#
  g0[1] = lds_byte_addr;                                   // lds_addr
  g0[2] = (unsigned)ga;                                    // global_addr[31:0]
  g0[3] = (unsigned)((ga >> 32) & 0x01FFFFFFu) | (2u << 30); // addr[56:32] | type=2
  v8i g1;
  g1[0] = (int)(1u << 16);                                 // data_size=1 (2 bytes)
  g1[1] = (int)((tdim0 & 0xFFFFu) << 16);                  // tensor_dim0[15:0] @bit48
  g1[2] = (int)((tdim0 >> 16) | ((tdim1 & 0xFFFFu) << 16));
  g1[3] = (int)((tdim1 >> 16) | (tile_x << 16));           // tile_dim0 @bit112
  g1[4] = (int)(tile_y | (tile_z << 16));                  // tile_dim1 | tile_dim2
  g1[5] = (int)(unsigned)(sy & 0xFFFFFFFFull);             // dim0_stride[31:0]
  g1[6] = (int)((unsigned)((sy >> 32) & 0xFFFFull) | ((unsigned)(sz & 0xFFFFull) << 16));
  g1[7] = (int)(unsigned)((sz >> 16) & 0xFFFFFFFFull);     // dim1_stride[47:16]
  v4i g2;
  g2[0] = (int)tdim2;                                      // tensor_dim2
  g2[1] = 0; g2[2] = 0; g2[3] = 0;
  v4i g3 = {0,0,0,0};
#if __clang_major__ >= 23
  v8i g4 = {0,0,0,0,0,0,0,0};
  __builtin_amdgcn_tensor_load_to_lds(g0, g1, g2, g3, g4, 0);
#else
  __builtin_amdgcn_tensor_load_to_lds(g0, g1, g2, g3, 0);
#endif
}
__device__ __forceinline__ unsigned lds_addr_of(const void* p){
  return (unsigned)(size_t)p;   // flat->LDS aperture: low 32 bits are the LDS byte offset
}
#endif

// A-fragment 16x32 f16 (row-major source): lane 0-15 = row M, lane-half picks K offset.
__device__ __forceinline__ v16h load_a_frag(const _Float16* base, int rstride, int m, int hl){
  v16h a;
  const _Float16* rp = base + (size_t)m * rstride;
#pragma unroll
  for (int v = 0; v < 8; ++v){
    int kk = (hl << 4) + (v << 1);
    a[2*v]   = rp[kk];
    a[2*v+1] = rp[kk+1];
  }
  return a;
}
// B-fragment 32x16 f16 from LDS column (K contiguous within column)
__device__ __forceinline__ v16h load_b_frag_lds(const _Float16* colp, int hl){
  v16h b;
#pragma unroll
  for (int v = 0; v < 8; ++v){
    int kk = (hl << 4) + (v << 1);
    b[2*v]   = colp[kk];
    b[2*v+1] = colp[kk+1];
  }
  return b;
}
// B-fragment with strided K (global, row stride = kstride elems); base includes N offset
__device__ __forceinline__ v16h load_b_frag_strided(const _Float16* base, int kstride, int hl){
  v16h b;
#pragma unroll
  for (int v = 0; v < 8; ++v){
    int kk = (hl << 4) + (v << 1);
    b[2*v]   = base[(size_t)kk * kstride];
    b[2*v+1] = base[(size_t)(kk+1) * kstride];
  }
  return b;
}

// ---------------- routing: pooled mean over padded length (1034) -> sigmoid(lin) --------
__global__ __launch_bounds__(256) void routing_kernel(
    const float* __restrict__ x, const float* __restrict__ rw,
    const float* __restrict__ rb, float* __restrict__ r4, int Cin)
{
  __shared__ float red[3][256];
  const int b = blockIdx.x, tid = threadIdx.x;
  float p0 = 0.f, p1 = 0.f, p2 = 0.f;
  for (int c = tid; c < Cin; c += 256){
    const float* xp = x + (((size_t)b * Cin + c) << 10);
    float s = 0.f;
    for (int l = 0; l < 1024; ++l) s += xp[l];
    s *= (1.f / 1034.f);
    p0 += s * rw[c]; p1 += s * rw[Cin + c]; p2 += s * rw[2*Cin + c];
  }
  red[0][tid] = p0; red[1][tid] = p1; red[2][tid] = p2;
  __syncthreads();
  for (int s = 128; s > 0; s >>= 1){
    if (tid < s){
      red[0][tid] += red[0][tid+s];
      red[1][tid] += red[1][tid+s];
      red[2][tid] += red[2][tid+s];
    }
    __syncthreads();
  }
  if (tid == 0){
    float r0 = sigmf(red[0][0] + rb[0]);
    float r1 = sigmf(red[1][0] + rb[1]);
    float r2 = sigmf(red[2][0] + rb[2]);
    r4[b*4+0] = r0; r4[b*4+1] = r1; r4[b*4+2] = r2; r4[b*4+3] = r0 + r1 + r2;
  }
}

// ---------------- weight packing -------------------------------------------------------
__global__ void pack_condw_kernel(const float* __restrict__ W, const float* __restrict__ r4,
                                  _Float16* __restrict__ Weff, int Cin, int total)
{
  int idx = blockIdx.x * 256 + threadIdx.x;
  if (idx >= total) return;
  int c = idx % Cin; int t = idx / Cin;
  int k = t % 11; t /= 11;
  int o = t % 256; int b = t / 256;
  size_t es = (size_t)256 * Cin * 11;
  size_t si = (((size_t)o * Cin + c) * 11) + k;
  float v = r4[b*4+0]*W[si] + r4[b*4+1]*W[es+si] + r4[b*4+2]*W[2*es+si];
  Weff[idx] = (_Float16)v;
}
__global__ void pack_gw_kernel(const float* __restrict__ W, _Float16* __restrict__ Wg,
                               int Cin, int total)
{
  int idx = blockIdx.x * 256 + threadIdx.x;
  if (idx >= total) return;
  int c = idx % Cin; int t = idx / Cin;
  int k = t % 11; int o = t / 11;
  Wg[idx] = (_Float16)W[(((size_t)o * Cin + c) * 11) + k];
}
__global__ void cvt_f16_kernel(const float* __restrict__ s, _Float16* __restrict__ d, int n){
  int i = blockIdx.x * 256 + threadIdx.x;
  if (i < n) d[i] = (_Float16)s[i];
}

// ---------------- gated cond-conv layer as fused implicit GEMM (WMMA + TDM) -------------
__global__ __launch_bounds__(256) void gated_conv_kernel(
    const _Float16* __restrict__ xf, const _Float16* __restrict__ Weff,
    const _Float16* __restrict__ Wg,
    const float* __restrict__ cond_b, const float* __restrict__ layer_b,
    const float* __restrict__ gate_b, const float* __restrict__ layer_c,
    const float* __restrict__ r4, const float* __restrict__ res,
    float* __restrict__ out, int Cin, int out_bstride, int add_res)
{
  __shared__ _Float16 xs[74][32];       // input tile, transposed (c contiguous)
  __shared__ _Float16 waC[64*11*32];    // cond weight tile [o][k][c] via TDM
  const int b  = blockIdx.z;
  const int l0 = blockIdx.x << 6;
  const int o0 = blockIdx.y << 6;
  const int tid  = threadIdx.x;
  const int lane = tid & 31, wv = tid >> 5;
  const int hl = lane >> 4, lm = lane & 15;
  const int mw = wv & 3, nw = wv >> 2;
  const int rstr = 11 * Cin;
  v8f accA0 = {0,0,0,0,0,0,0,0};
  v8f accA1 = accA0, accG0 = accA0, accG1 = accA0;
  const int NC = Cin >> 5;
  for (int cb = 0; cb < NC; ++cb){
    const int c0 = cb << 5;
    __syncthreads();
#if HAVE_TDM
    if (wv == 0){
      tdm_load_tile(lds_addr_of(waC),
                    Weff + (((size_t)(b*256 + o0)) * 11) * Cin + c0,
                    32u, 11u, 64u,
                    (unsigned long long)Cin, (unsigned long long)rstr,
                    (unsigned)Cin, 11u, 256u);
    }
#else
    for (int idx = tid; idx < 64*11*32; idx += 256){
      int o = idx / 352; int r = idx - o * 352;
      int k = r >> 5; int c = r & 31;
      waC[idx] = Weff[(((size_t)(b*256 + o0 + o)) * 11 + k) * Cin + c0 + c];
    }
#endif
    for (int idx = tid; idx < 74*32; idx += 256){
      int cc = idx / 74, col = idx - cc * 74;
      int l = l0 - 10 + col;
      _Float16 v = (_Float16)0.f;
      if (l >= 0) v = xf[(((size_t)b * Cin + c0 + cc) << 10) + l];
      xs[col][cc] = v;
      if (cb + 1 < NC && l >= 0)
        __builtin_prefetch(xf + (((size_t)b * Cin + c0 + 32 + cc) << 10) + l, 0, 0);
    }
#if HAVE_TDM
    if (wv == 0) __builtin_amdgcn_s_wait_tensorcnt(0);
#endif
    __syncthreads();
    for (int k = 0; k < 11; ++k){
      const _Float16* aCb = &waC[((mw*16)*11 + k) * 32];
      const _Float16* aGb = Wg + (((size_t)(o0 + mw*16)) * 11 + k) * Cin + c0;
      v16h aC = load_a_frag(aCb, 352, lm, hl);
      v16h aG = load_a_frag(aGb, rstr, lm, hl);
      {
        int col = nw*32 + lm + k;
        v16h bf = load_b_frag_lds(&xs[col][0], hl);
        accA0 = WMMA_F16(aC, bf, accA0);
        accG0 = WMMA_F16(aG, bf, accG0);
      }
      {
        int col = nw*32 + 16 + lm + k;
        v16h bf = load_b_frag_lds(&xs[col][0], hl);
        accA1 = WMMA_F16(aC, bf, accA1);
        accG1 = WMMA_F16(aG, bf, accG1);
      }
    }
  }
  const float rsum = r4[b*4+3];
#pragma unroll
  for (int nn = 0; nn < 2; ++nn){
    v8f aA = nn ? accA1 : accA0;
    v8f aG = nn ? accG1 : accG0;
#pragma unroll
    for (int rg = 0; rg < 8; ++rg){
      int o = o0 + mw*16 + rg + (hl << 3);
      int l = l0 + nw*32 + nn*16 + lm;
      float va = aA[rg] + rsum * cond_b[o] + layer_b[o];
      float vg = aG[rg] + gate_b[o] + layer_c[o];
      float hv = va * sigmf(vg);
      if (add_res) hv += res[(((size_t)b << 8) + o) * 1024 + l];
      out[(size_t)b * out_bstride + ((size_t)o << 10) + l] = hv;
    }
  }
}

// ---------------- LSTM packing ----------------------------------------------------------
__global__ void pack_seqT_kernel(const float* __restrict__ seq, _Float16* __restrict__ st){
  int i = blockIdx.x * 256 + threadIdx.x;
  if (i >= 2097152) return;
  int e = i & 127; int t = (i >> 7) & 1023; int b = i >> 17;
  st[i] = (_Float16)seq[(((size_t)(b << 7) + e) << 10) + t];
}
__global__ void pack_wihT_kernel(const float* __restrict__ w, _Float16* __restrict__ d){
  int i = blockIdx.x * 256 + threadIdx.x;
  if (i >= 262144) return;
  int j = i & 1023; int e = (i >> 10) & 127; int dd = i >> 17;
  d[i] = (_Float16)w[(size_t)dd * 131072 + (size_t)j * 128 + e];
}
__global__ void pack_whhT_kernel(const float* __restrict__ w, _Float16* __restrict__ d){
  int i = blockIdx.x * 256 + threadIdx.x;
  if (i >= 524288) return;
  int j = i & 1023; int k = (i >> 10) & 255; int dd = i >> 18;
  d[i] = (_Float16)w[(size_t)dd * 262144 + (size_t)j * 256 + k];
}
__global__ void pack_bsum_kernel(const float* __restrict__ a, const float* __restrict__ b,
                                 float* __restrict__ d){
  int i = blockIdx.x * 256 + threadIdx.x;
  if (i < 2048) d[i] = a[i] + b[i];
}

// gin[b,t,j] = seqT[b,t,:] @ WihT[:,j] + bsum[j]     (M=t, N=j, K=128)
__global__ __launch_bounds__(256) void gin_gemm_kernel(
    const _Float16* __restrict__ seqT, const _Float16* __restrict__ wihT,
    const float* __restrict__ bsum, float* __restrict__ gin)
{
  const int b  = blockIdx.z;
  const int t0 = blockIdx.x << 6;
  const int j0 = blockIdx.y << 6;
  const int tid = threadIdx.x;
  const int lane = tid & 31, wv = tid >> 5;
  const int hl = lane >> 4, lm = lane & 15;
  const int mw = wv & 3, nw = wv >> 2;
  v8f acc0 = {0,0,0,0,0,0,0,0};
  v8f acc1 = acc0;
  for (int e0 = 0; e0 < 128; e0 += 32){
    const _Float16* ab = seqT + ((size_t)((b << 10) + t0 + mw*16)) * 128 + e0;
    v16h a = load_a_frag(ab, 128, lm, hl);
    const _Float16* bb = wihT + ((size_t)e0 << 10) + j0 + nw*32 + lm;
    v16h b0 = load_b_frag_strided(bb, 1024, hl);
    acc0 = WMMA_F16(a, b0, acc0);
    v16h b1 = load_b_frag_strided(bb + 16, 1024, hl);
    acc1 = WMMA_F16(a, b1, acc1);
  }
#pragma unroll
  for (int rg = 0; rg < 8; ++rg){
    int t = t0 + mw*16 + rg + (hl << 3);
    int j = j0 + nw*32 + lm;
    size_t base = ((size_t)((b << 10) + t)) * 1024;
    gin[base + j]      = acc0[rg] + bsum[j];
    gin[base + j + 16] = acc1[rg] + bsum[j + 16];
  }
}

// persistent recurrent LSTM: one workgroup, state in LDS, WMMA per step (M=16 batch rows)
__global__ __launch_bounds__(256) void lstm_rec_kernel(
    const float* __restrict__ gin, const _Float16* __restrict__ whhT,
    float* __restrict__ hcat, int coff, int rev)
{
  __shared__ _Float16 hh[16][256];
  __shared__ float    cst[16][256];
  __shared__ _Float16 gbuf[16][1024];
  const int tid = threadIdx.x;
  const int lane = tid & 31, wv = tid >> 5;
  const int hl = lane >> 4, lm = lane & 15;
  for (int i = tid; i < 4096; i += 256){ hh[i >> 8][i & 255] = (_Float16)0.f; cst[i >> 8][i & 255] = 0.f; }
  __syncthreads();
  for (int step = 0; step < 1024; ++step){
    const int t = rev ? (1023 - step) : step;
#pragma unroll 1
    for (int it = 0; it < 8; ++it){
      const int j0 = ((wv << 3) + it) << 4;
      v8f acc = {0,0,0,0,0,0,0,0};
      for (int k0 = 0; k0 < 256; k0 += 32){
        v16h a  = load_a_frag(&hh[0][0] + k0, 256, lm, hl);
        v16h bf = load_b_frag_strided(whhT + ((size_t)k0 << 10) + j0 + lm, 1024, hl);
        acc = WMMA_F16(a, bf, acc);
      }
#pragma unroll
      for (int rg = 0; rg < 8; ++rg){
        int bb = rg + (hl << 3);
        int j  = j0 + lm;
        gbuf[bb][j] = (_Float16)(acc[rg] + gin[((size_t)((bb << 10) + t)) * 1024 + j]);
      }
    }
    __syncthreads();
    for (int i = tid; i < 4096; i += 256){
      int bb = i >> 8, hx = i & 255;
      float gi = (float)gbuf[bb][hx];
      float gf = (float)gbuf[bb][256 + hx];
      float gg = (float)gbuf[bb][512 + hx];
      float go = (float)gbuf[bb][768 + hx];
      float c  = sigmf(gf) * cst[bb][hx] + sigmf(gi) * tanhf(gg);
      float hv = sigmf(go) * tanhf(c);
      cst[bb][hx] = c; hh[bb][hx] = (_Float16)hv;
      hcat[((size_t)bb * 768 + coff + hx) * 1024 + t] = hv;
    }
    __syncthreads();
  }
}

// ---------------- ASPP branch: dilated 3-tap conv + ReLU + BN (WMMA + TDM) --------------
__global__ void pack_asppW_kernel(const float* __restrict__ W, _Float16* __restrict__ Wp){
  int i = blockIdx.x * 256 + threadIdx.x;
  if (i >= 1769472) return;
  int c = i % 768; int t = i / 768;
  int w = t % 3; int o = t / 3;
  Wp[i] = (_Float16)W[(((size_t)o * 768 + c) * 3) + w];
}
__global__ __launch_bounds__(256) void aspp_kernel(
    const _Float16* __restrict__ hf, const _Float16* __restrict__ Wp,
    const float* __restrict__ ab, const float* __restrict__ bng,
    const float* __restrict__ bnb, float* __restrict__ y, int dil)
{
  __shared__ _Float16 xs[80][32];
  __shared__ _Float16 wa[64*3*32];      // weight tile [o][w][c] via TDM
  const int b  = blockIdx.z;
  const int l0 = blockIdx.x << 6;
  const int o0 = blockIdx.y << 6;
  const int tid = threadIdx.x;
  const int lane = tid & 31, wv = tid >> 5;
  const int hl = lane >> 4, lm = lane & 15;
  const int mw = wv & 3, nw = wv >> 2;
  const int CW = 64 + 2 * dil;
  v8f acc0 = {0,0,0,0,0,0,0,0};
  v8f acc1 = acc0;
  for (int cb = 0; cb < 24; ++cb){
    const int c0 = cb << 5;
    __syncthreads();
#if HAVE_TDM
    if (wv == 0){
      tdm_load_tile(lds_addr_of(wa),
                    Wp + ((size_t)o0 * 3) * 768 + c0,
                    32u, 3u, 64u,
                    768ull, 2304ull, 768u, 3u, 768u);
    }
#else
    for (int idx = tid; idx < 64*3*32; idx += 256){
      int o = idx / 96; int r = idx - o * 96;
      int w = r >> 5; int c = r & 31;
      wa[idx] = Wp[(((size_t)(o0 + o)) * 3 + w) * 768 + c0 + c];
    }
#endif
    for (int idx = tid; idx < CW * 32; idx += 256){
      int cc = idx / CW, col = idx - cc * CW;
      int l = l0 - dil + col;
      _Float16 v = (_Float16)0.f;
      if (l >= 0 && l < 1024) v = hf[(((size_t)b * 768 + c0 + cc) << 10) + l];
      xs[col][cc] = v;
    }
#if HAVE_TDM
    if (wv == 0) __builtin_amdgcn_s_wait_tensorcnt(0);
#endif
    __syncthreads();
    for (int w = 0; w < 3; ++w){
      const _Float16* abse = &wa[((mw*16)*3 + w) * 32];
      v16h a = load_a_frag(abse, 96, lm, hl);
      int colb = nw*32 + lm + w * dil;
      v16h b0 = load_b_frag_lds(&xs[colb][0], hl);
      acc0 = WMMA_F16(a, b0, acc0);
      v16h b1 = load_b_frag_lds(&xs[colb + 16][0], hl);
      acc1 = WMMA_F16(a, b1, acc1);
    }
  }
  const float inv_std = 0.9999950000374997f;   // 1/sqrt(1+1e-5)
#pragma unroll
  for (int rg = 0; rg < 8; ++rg){
    int o = o0 + mw*16 + rg + (hl << 3);
    int l = l0 + nw*32 + lm;
    float v0 = acc0[rg] + ab[o]; v0 = v0 > 0.f ? v0 : 0.f; v0 = v0 * inv_std * bng[o] + bnb[o];
    float v1 = acc1[rg] + ab[o]; v1 = v1 > 0.f ? v1 : 0.f; v1 = v1 * inv_std * bng[o] + bnb[o];
    size_t base = (((size_t)b * 768 + o) << 10) + l;
    y[base]      = v0;
    y[base + 16] = v1;
  }
}

// ---------------- head: 1x1 conv group accumulation + logsigmoid ------------------------
__global__ __launch_bounds__(256) void head_acc_kernel(
    const float* __restrict__ feat, const float* __restrict__ w1,
    float* __restrict__ acc, int init)
{
  __shared__ float ws[6912];           // 9 x 768
  const int tid = threadIdx.x;
  for (int i = tid; i < 6912; i += 256){
    int cls = i / 768, c = i - cls * 768;
    ws[i] = w1[(size_t)cls * 3840 + c];
  }
  __syncthreads();
  const int l = blockIdx.x * 256 + tid;
  const int b = blockIdx.y;
  float a[9];
#pragma unroll
  for (int cls = 0; cls < 9; ++cls) a[cls] = 0.f;
  for (int c = 0; c < 768; ++c){
    float f = feat[(((size_t)b * 768 + c) << 10) + l];
#pragma unroll
    for (int cls = 0; cls < 9; ++cls) a[cls] += ws[cls * 768 + c] * f;
  }
#pragma unroll
  for (int cls = 0; cls < 9; ++cls){
    size_t oi = (((size_t)b * 9 + cls) << 10) + l;
    if (init) acc[oi] = a[cls];
    else      acc[oi] += a[cls];
  }
}
__global__ void logsig_kernel(const float* __restrict__ acc, const float* __restrict__ b9,
                              float* __restrict__ out)
{
  int i = blockIdx.x * 256 + threadIdx.x;
  if (i >= 147456) return;
  int cls = (i >> 10) % 9;
  float x = acc[i] + b9[cls];
  out[i] = fminf(x, 0.f) - log1pf(__expf(-fabsf(x)));
}

// ---------------- workspace layout (phase-aliased) --------------------------------------
static constexpr size_t O_XF16 = 0;          // 8 MB   f16 conv input
static constexpr size_t O_H0   = 8388608;    // 16 MB  f32
static constexpr size_t O_H1   = 25165824;   // 16 MB
static constexpr size_t O_H2   = 41943040;   // 16 MB
static constexpr size_t O_WEFF = 58720256;   // 23 MB  f16 [16][256][11][256]
static constexpr size_t O_WGT  = 81788928;   // 1.4 MB f16 [256][11][256]
static constexpr size_t O_R4   = 83230720;   // 256 B
static constexpr size_t O_HCAT = 83230976;   // 50.3 MB f32 [16][768][1024]
// LSTM phase (aliases conv region, which is dead by then)
static constexpr size_t O_GIN  = 0;          // 64 MB  f32 [16][1024][1024]
static constexpr size_t O_SEQT = 67108864;   // 4 MB   f16 [16][1024][128]
static constexpr size_t O_WIHT = 71303168;   // 0.5 MB
static constexpr size_t O_WHHT = 71827456;   // 1 MB
static constexpr size_t O_BSUM = 72876032;   // 8 KB
// ASPP phase (aliases conv/LSTM region)
static constexpr size_t O_HF16 = 0;          // 25 MB  f16 hcat
static constexpr size_t O_WP   = 25165824;   // 3.5 MB f16 [768][3][768]
static constexpr size_t O_Y    = 28704768;   // 50.3 MB f32
static constexpr size_t O_ACC  = 79036416;   // 0.6 MB f32 [16][9][1024]

extern "C" void kernel_launch(void* const* d_in, const int* in_sizes, int n_in,
                              void* d_out, int out_size, void* d_ws, size_t ws_size,
                              hipStream_t stream)
{
  (void)in_sizes; (void)n_in; (void)out_size; (void)ws_size;
  const float* seq      = (const float*)d_in[0];
  const float* cond0_w  = (const float*)d_in[1];
  const float* cond0_b  = (const float*)d_in[2];
  const float* rout0_w  = (const float*)d_in[3];
  const float* rout0_b  = (const float*)d_in[4];
  const float* b0       = (const float*)d_in[5];
  const float* gate0_w  = (const float*)d_in[6];
  const float* gate0_b  = (const float*)d_in[7];
  const float* c0_      = (const float*)d_in[8];
  const float* cond_ws  = (const float*)d_in[9];
  const float* cond_bs  = (const float*)d_in[10];
  const float* rout_ws  = (const float*)d_in[11];
  const float* rout_bs  = (const float*)d_in[12];
  const float* bs_      = (const float*)d_in[13];
  const float* gate_ws  = (const float*)d_in[14];
  const float* gate_bs  = (const float*)d_in[15];
  const float* cs_      = (const float*)d_in[16];
  const float* lstm_wih = (const float*)d_in[17];
  const float* lstm_whh = (const float*)d_in[18];
  const float* lstm_bih = (const float*)d_in[19];
  const float* lstm_bhh = (const float*)d_in[20];
  const float* aspp_ws  = (const float*)d_in[21];
  const float* aspp_bs  = (const float*)d_in[22];
  const float* bn_g     = (const float*)d_in[23];
  const float* bn_b     = (const float*)d_in[24];
  const float* conv1_w  = (const float*)d_in[25];
  const float* conv1_b  = (const float*)d_in[26];

  char* W = (char*)d_ws;
  _Float16* xf16 = (_Float16*)(W + O_XF16);
  float*    H0   = (float*)(W + O_H0);
  float*    H1   = (float*)(W + O_H1);
  float*    H2   = (float*)(W + O_H2);
  _Float16* Weff = (_Float16*)(W + O_WEFF);
  _Float16* Wg   = (_Float16*)(W + O_WGT);
  float*    r4   = (float*)(W + O_R4);
  float*    hcat = (float*)(W + O_HCAT);
  float*    gin  = (float*)(W + O_GIN);
  _Float16* seqT = (_Float16*)(W + O_SEQT);
  _Float16* wihT = (_Float16*)(W + O_WIHT);
  _Float16* whhT = (_Float16*)(W + O_WHHT);
  float*    bsum = (float*)(W + O_BSUM);
  _Float16* hf16 = (_Float16*)(W + O_HF16);
  _Float16* Wp   = (_Float16*)(W + O_WP);
  float*    ybuf = (float*)(W + O_Y);
  float*    accb = (float*)(W + O_ACC);

  // ---- layer 0 (Cin = 128) ----
  routing_kernel<<<16, 256, 0, stream>>>(seq, rout0_w, rout0_b, r4, 128);
  pack_condw_kernel<<<22528, 256, 0, stream>>>(cond0_w, r4, Weff, 128, 5767168);
  pack_gw_kernel<<<1408, 256, 0, stream>>>(gate0_w, Wg, 128, 360448);
  cvt_f16_kernel<<<8192, 256, 0, stream>>>(seq, xf16, 2097152);
  gated_conv_kernel<<<dim3(16, 4, 16), 256, 0, stream>>>(
      xf16, Weff, Wg, cond0_b, b0, gate0_b, c0_, r4, nullptr, H0, 128, 262144, 0);

  // ---- 4 residual layers (Cin = 256); i=3 writes straight into hcat channels 0..255 ----
  const float* hin[4]  = {H0, H1, H2, H0};
  float*       hout[4] = {H1, H2, H0, hcat};
  const float* resv[4] = {H0, nullptr, H1, nullptr};
  const int addres[4]  = {1, 0, 1, 0};
  const int obst[4]    = {262144, 262144, 262144, 786432};
  for (int i = 0; i < 4; ++i){
    routing_kernel<<<16, 256, 0, stream>>>(hin[i], rout_ws + (size_t)i*768, rout_bs + i*3, r4, 256);
    pack_condw_kernel<<<45056, 256, 0, stream>>>(cond_ws + (size_t)i*2162688, r4, Weff, 256, 11534336);
    pack_gw_kernel<<<2816, 256, 0, stream>>>(gate_ws + (size_t)i*720896, Wg, 256, 720896);
    cvt_f16_kernel<<<16384, 256, 0, stream>>>(hin[i], xf16, 4194304);
    gated_conv_kernel<<<dim3(16, 4, 16), 256, 0, stream>>>(
        xf16, Weff, Wg, cond_bs + (size_t)i*256, bs_ + (size_t)i*256,
        gate_bs + (size_t)i*256, cs_ + (size_t)i*256, r4, resv[i], hout[i],
        256, obst[i], addres[i]);
  }

  // ---- biLSTM ----
  pack_seqT_kernel<<<8192, 256, 0, stream>>>(seq, seqT);
  pack_wihT_kernel<<<1024, 256, 0, stream>>>(lstm_wih, wihT);
  pack_whhT_kernel<<<2048, 256, 0, stream>>>(lstm_whh, whhT);
  pack_bsum_kernel<<<8, 256, 0, stream>>>(lstm_bih, lstm_bhh, bsum);
  for (int d = 0; d < 2; ++d){
    gin_gemm_kernel<<<dim3(16, 16, 16), 256, 0, stream>>>(
        seqT, wihT + (size_t)d*131072, bsum + (size_t)d*1024, gin);
    lstm_rec_kernel<<<1, 256, 0, stream>>>(
        gin, whhT + (size_t)d*262144, hcat, 256 + d*256, d);
  }

  // ---- head group 0 (identity feature) + ASPP branches ----
  head_acc_kernel<<<dim3(4, 16), 256, 0, stream>>>(hcat, conv1_w, accb, 1);
  cvt_f16_kernel<<<49152, 256, 0, stream>>>(hcat, hf16, 12582912);
  const int dils[4] = {1, 2, 4, 8};
  for (int j = 0; j < 4; ++j){
    pack_asppW_kernel<<<6912, 256, 0, stream>>>(aspp_ws + (size_t)j*1769472, Wp);
    aspp_kernel<<<dim3(16, 12, 16), 256, 0, stream>>>(
        hf16, Wp, aspp_bs + (size_t)j*768, bn_g + (size_t)j*768, bn_b + (size_t)j*768,
        ybuf, dils[j]);
    head_acc_kernel<<<dim3(4, 16), 256, 0, stream>>>(ybuf, conv1_w + (size_t)(j+1)*768, accb, 0);
  }
  logsig_kernel<<<576, 256, 0, stream>>>(accb, conv1_b, (float*)d_out);
}
